// SVDQuantLinear_56727928045732
// MI455X (gfx1250) — 8-bit matrix operand order repaired and checked by reading, compiled, not run on hardware
//
#include <hip/hip_runtime.h>
#include <stdint.h>

typedef int   v8i __attribute__((ext_vector_type(8)));
typedef float v8f __attribute__((ext_vector_type(8)));
typedef float v2f __attribute__((ext_vector_type(2)));

constexpr int MDIM = 8192;
constexpr int KDIM = 4096;
constexpr int NDIM = 4096;
constexpr int RDIM = 32;
constexpr int GROUPS = KDIM / 64;

__global__ __launch_bounds__(256) void quant_rows_kernel(
    const float* __restrict__ src, const float* __restrict__ smooth,
    uint8_t* __restrict__ qout, float* __restrict__ scalesT, int rows)
{
  const int wave = threadIdx.x >> 5;
  const int lane = threadIdx.x & 31;
  const int row  = blockIdx.x * 8 + wave;

  const float* s = src + (size_t)row * KDIM;
  uint32_t* q32  = (uint32_t*)(qout + (size_t)row * KDIM);

  #pragma unroll 1
  for (int it = 0; it < KDIM / 128; ++it) {
    const int k = it * 128 + lane * 4;
    float4 v = *(const float4*)(s + k);
    if (smooth) {
      float4 sm = *(const float4*)(smooth + k);
      v.x *= sm.x; v.y *= sm.y; v.z *= sm.z; v.w *= sm.w;
    }
    float am = fmaxf(fmaxf(fabsf(v.x), fabsf(v.y)), fmaxf(fabsf(v.z), fabsf(v.w)));
    am = fmaxf(am, __shfl_xor(am, 1, 32));
    am = fmaxf(am, __shfl_xor(am, 2, 32));
    am = fmaxf(am, __shfl_xor(am, 4, 32));
    am = fmaxf(am, __shfl_xor(am, 8, 32));
    const float scale = fmaxf(am * (1.0f / 7.0f), 1e-8f);
    const float inv   = 1.0f / scale;

    float r0 = fminf(7.0f, fmaxf(-8.0f, rintf(v.x * inv)));
    float r1 = fminf(7.0f, fmaxf(-8.0f, rintf(v.y * inv)));
    float r2 = fminf(7.0f, fmaxf(-8.0f, rintf(v.z * inv)));
    float r3 = fminf(7.0f, fmaxf(-8.0f, rintf(v.w * inv)));
    int q0 = (int)r0, q1 = (int)r1, q2 = (int)r2, q3 = (int)r3;
    uint32_t p = (uint32_t)(q0 & 0xFF)        | ((uint32_t)(q1 & 0xFF) << 8) |
                 ((uint32_t)(q2 & 0xFF) << 16) | ((uint32_t)(q3 & 0xFF) << 24);
    q32[k >> 2] = p;

    if ((lane & 15) == 0) {
      const int g = it * 2 + (lane >> 4);
      scalesT[(size_t)g * rows + row] = scale;
    }
  }
}

__global__ __launch_bounds__(256) void lora_down_kernel(
    const float* __restrict__ x, const float* __restrict__ smooth,
    const float* __restrict__ down, float* __restrict__ t)
{
  const int wave = threadIdx.x >> 5;
  const int r    = threadIdx.x & 31;
  const int row  = blockIdx.x * 8 + wave;
  const float* xr = x + (size_t)row * KDIM;

  float acc = 0.0f;
  #pragma unroll 4
  for (int k = 0; k < KDIM; ++k) {
    acc = fmaf(xr[k] * smooth[k], down[(size_t)k * RDIM + r], acc);
  }
  t[(size_t)row * RDIM + r] = acc;
}

__device__ __forceinline__ void consume_group(v8f& acc, const v8i& di,
                                              const float4 sxl, const float4 sxh,
                                              const float swv) {
  acc[0] += (float)di[0] * (sxl.x * swv);
  acc[1] += (float)di[1] * (sxl.y * swv);
  acc[2] += (float)di[2] * (sxl.z * swv);
  acc[3] += (float)di[3] * (sxl.w * swv);
  acc[4] += (float)di[4] * (sxh.x * swv);
  acc[5] += (float)di[5] * (sxh.y * swv);
  acc[6] += (float)di[6] * (sxh.z * swv);
  acc[7] += (float)di[7] * (sxh.w * swv);
}

__global__ __launch_bounds__(256) void w4a4_gemm_kernel(
    const uint8_t* __restrict__ qx, const float* __restrict__ sxT,
    const uint8_t* __restrict__ qw, const float* __restrict__ swT,
    const float* __restrict__ t,  const float* __restrict__ up,
    const float* __restrict__ bias, float* __restrict__ out)
{
  const int wave = threadIdx.x >> 5;
  const int lane = threadIdx.x & 31;
  const int h    = lane >> 4;
  const int ln   = lane & 15;

  const int m0 = blockIdx.y * 64 + (wave >> 2) * 32;
  const int n0 = blockIdx.x * 128 + (wave & 3) * 32;

  const int hb = h * 16;
  const uint8_t* arow0 = qx + (size_t)(m0 + ln) * KDIM;
  const uint8_t* arow1 = qx + (size_t)(m0 + 16 + ln) * KDIM;
  const uint8_t* brow0 = qw + (size_t)(n0 + ln) * KDIM + h * 16;
  const uint8_t* brow1 = qw + (size_t)(n0 + 16 + ln) * KDIM + h * 16;

  auto loadA = [&](const uint8_t* ar, int g) -> v8i {
    const uint8_t* ab = ar + g * 64;
    v8i a;
    a[0] = *(const int*)(ab + 0 + hb);  a[1] = *(const int*)(ab + 4 + hb);
    a[2] = *(const int*)(ab + 8 + hb);  a[3] = *(const int*)(ab + 12 + hb);
    a[4] = *(const int*)(ab + 32 + hb); a[5] = *(const int*)(ab + 36 + hb);
    a[6] = *(const int*)(ab + 40 + hb); a[7] = *(const int*)(ab + 44 + hb);
    return a;
  };
  auto loadB = [&](const uint8_t* br, int g) -> v8i {
    uint4 lo = *(const uint4*)(br + g * 64);
    uint4 hi = *(const uint4*)(br + g * 64 + 32);
    v8i b;
    b[0] = (int)lo.x; b[1] = (int)lo.y; b[2] = (int)lo.z; b[3] = (int)lo.w;
    b[4] = (int)hi.x; b[5] = (int)hi.y; b[6] = (int)hi.z; b[7] = (int)hi.w;
    return b;
  };
  auto wmma8 = [](const v8i& a, const v8i& b) -> v8i {
    v8i czero = {};
    return __builtin_amdgcn_wmma_i32_16x16x64_iu8(true, a, true, b, czero,
                                                  false, false);
  };

  v8f acc00 = {}, acc01 = {}, acc10 = {}, acc11 = {};

  v8i a0 = loadA(arow0, 0), a1 = loadA(arow1, 0);
  v8i b0 = loadB(brow0, 0), b1 = loadB(brow1, 0);
  v8i d00p = wmma8(a0, b0);
  v8i d01p = wmma8(a0, b1);
  v8i d10p = wmma8(a1, b0);
  v8i d11p = wmma8(a1, b1);
  const float* sxg = sxT + m0 + 8 * h;
  float4 sxl0_p = *(const float4*)(sxg);
  float4 sxh0_p = *(const float4*)(sxg + 4);
  float4 sxl1_p = *(const float4*)(sxg + 16);
  float4 sxh1_p = *(const float4*)(sxg + 20);
  float  sw0_p  = swT[n0 + ln];
  float  sw1_p  = swT[n0 + 16 + ln];

  #pragma unroll 2
  for (int g = 1; g < GROUPS; ++g) {
    __builtin_prefetch(brow0 + (g + 1) * 64, 0, 1);

    a0 = loadA(arow0, g); a1 = loadA(arow1, g);
    b0 = loadB(brow0, g); b1 = loadB(brow1, g);
    v8i d00 = wmma8(a0, b0);
    v8i d01 = wmma8(a0, b1);
    v8i d10 = wmma8(a1, b0);
    v8i d11 = wmma8(a1, b1);

    consume_group(acc00, d00p, sxl0_p, sxh0_p, sw0_p);
    consume_group(acc01, d01p, sxl0_p, sxh0_p, sw1_p);
    consume_group(acc10, d10p, sxl1_p, sxh1_p, sw0_p);
    consume_group(acc11, d11p, sxl1_p, sxh1_p, sw1_p);

    d00p = d00; d01p = d01; d10p = d10; d11p = d11;
    const float* sxg2 = sxT + (size_t)g * MDIM + m0 + 8 * h;
    sxl0_p = *(const float4*)(sxg2);
    sxh0_p = *(const float4*)(sxg2 + 4);
    sxl1_p = *(const float4*)(sxg2 + 16);
    sxh1_p = *(const float4*)(sxg2 + 20);
    sw0_p  = swT[(size_t)g * NDIM + n0 + ln];
    sw1_p  = swT[(size_t)g * NDIM + n0 + 16 + ln];
  }

  consume_group(acc00, d00p, sxl0_p, sxh0_p, sw0_p);
  consume_group(acc01, d01p, sxl0_p, sxh0_p, sw1_p);
  consume_group(acc10, d10p, sxl1_p, sxh1_p, sw0_p);
  consume_group(acc11, d11p, sxl1_p, sxh1_p, sw1_p);

  const float* trow0 = t  + (size_t)(m0 + ln) * RDIM;
  const float* trow1 = t  + (size_t)(m0 + 16 + ln) * RDIM;
  const float* urow0 = up + (size_t)(n0 + ln) * RDIM;
  const float* urow1 = up + (size_t)(n0 + 16 + ln) * RDIM;
  #pragma unroll
  for (int s = 0; s < RDIM / 4; ++s) {
    const int k0 = s * 4;
    v2f av0 = { trow0[k0 + 2 * h], trow0[k0 + 1 + 2 * h] };
    v2f av1 = { trow1[k0 + 2 * h], trow1[k0 + 1 + 2 * h] };
    v2f bv0 = { urow0[k0 + 2 * h], urow0[k0 + 1 + 2 * h] };
    v2f bv1 = { urow1[k0 + 2 * h], urow1[k0 + 1 + 2 * h] };
    acc00 = __builtin_amdgcn_wmma_f32_16x16x4_f32(false, av0, false, bv0,
                                                  (short)0, acc00, false, false);
    acc01 = __builtin_amdgcn_wmma_f32_16x16x4_f32(false, av0, false, bv1,
                                                  (short)0, acc01, false, false);
    acc10 = __builtin_amdgcn_wmma_f32_16x16x4_f32(false, av1, false, bv0,
                                                  (short)0, acc10, false, false);
    acc11 = __builtin_amdgcn_wmma_f32_16x16x4_f32(false, av1, false, bv1,
                                                  (short)0, acc11, false, false);
  }

  const float bn0 = bias[n0 + ln];
  const float bn1 = bias[n0 + 16 + ln];
  #pragma unroll
  for (int i = 0; i < 8; ++i) {
    const size_t r0 = (size_t)(m0 + i + 8 * h) * NDIM;
    const size_t r1 = (size_t)(m0 + 16 + i + 8 * h) * NDIM;
    out[r0 + n0 + ln]      = acc00[i] + bn0;
    out[r0 + n0 + 16 + ln] = acc01[i] + bn1;
    out[r1 + n0 + ln]      = acc10[i] + bn0;
    out[r1 + n0 + 16 + ln] = acc11[i] + bn1;
  }
}

extern "C" void kernel_launch(void* const* d_in, const int* in_sizes, int n_in,
                              void* d_out, int out_size, void* d_ws, size_t ws_size,
                              hipStream_t stream) {
  const float* x      = (const float*)d_in[0];
  const float* w_res  = (const float*)d_in[1];
  const float* down   = (const float*)d_in[2];
  const float* up     = (const float*)d_in[3];
  const float* smooth = (const float*)d_in[4];
  const float* bias   = (const float*)d_in[5];
  float* out = (float*)d_out;

  uint8_t* ws = (uint8_t*)d_ws;
  size_t o = 0;
  uint8_t* qx = ws + o; o += (size_t)MDIM * KDIM;
  uint8_t* qw = ws + o; o += (size_t)NDIM * KDIM;
  float*   sxT = (float*)(ws + o); o += (size_t)GROUPS * MDIM * 4;
  float*   swT = (float*)(ws + o); o += (size_t)GROUPS * NDIM * 4;
  float*   t   = (float*)(ws + o); o += (size_t)MDIM * RDIM * 4;

  quant_rows_kernel<<<MDIM / 8, 256, 0, stream>>>(x, smooth, qx, sxT, MDIM);
  quant_rows_kernel<<<NDIM / 8, 256, 0, stream>>>(w_res, nullptr, qw, swT, NDIM);
  lora_down_kernel<<<MDIM / 8, 256, 0, stream>>>(x, smooth, down, t);

  dim3 grid(NDIM / 128, MDIM / 64);
  w4a4_gemm_kernel<<<grid, 256, 0, stream>>>(qx, sxT, qw, swT, t, up, bias, out);
}
